// BILSTM_withAttention2layer_26293789786982
// MI455X (gfx1250) — matrix operand order repaired and checked by reading, not compiled, not run on hardware
//
#include <hip/hip_runtime.h>

// ---------------------------------------------------------------------------
// BiLSTM(2-layer) + attention for MI455X (gfx1250, wave32, WMMA).
// Persistent-workgroup LSTM: Whh resident in LDS (f16, WMMA B-operand
// swizzled), cell state in VGPR accumulators, v_wmma_f32_16x16x32_f16
// for every matmul, LDS-only split barriers on the recurrent critical path.
// Attention GEMMs tiled 64x64 through LDS with async global->LDS staging.
// ---------------------------------------------------------------------------

#define TT 512   // sequence length
#define BB 64    // batch
#define HH 128   // hidden per direction
#define GG 512   // 4*H gates
#define DD 256   // 2*H bidirectional width
#define EE 10    // embedding dim

typedef __attribute__((ext_vector_type(16))) _Float16 v16h;
typedef __attribute__((ext_vector_type(8)))  float    v8f;

__device__ __forceinline__ float sigm(float x) { return 1.0f / (1.0f + __expf(-x)); }

// tanh(x) = 1 - 2/(exp(2x)+1): one v_exp_f32 + rcp, saturates correctly.
__device__ __forceinline__ float fast_tanh(float x) {
  float e = __expf(2.0f * x);
  return 1.0f - 2.0f / (e + 1.0f);
}

__device__ __forceinline__ v8f wmma16(v16h a, v16h b, v8f c) {
  // D = A(16x32 f16) * B(32x16 f16) + C(16x16 f32)
  return __builtin_amdgcn_wmma_f32_16x16x32_f16(false, a, false, b, (short)0, c,
                                                false, false);
}

// K' index inside a 32-wide chunk for A-operand element j, lane-half hi.
__device__ __forceinline__ int akidx(int hi, int j) {
  return (j < 8) ? (hi * 8 + j) : (16 + hi * 8 + (j - 8));
}

// Async DWORD copy global -> LDS (per-lane scatter). Low 32 bits of a flat
// LDS pointer are the LDS address (ISA 10.2 aperture rules).
__device__ __forceinline__ void async_g2l_b32(const void* g, void* l) {
  unsigned ldsoff = (unsigned)(unsigned long long)l;
  asm volatile("global_load_async_to_lds_b32 %0, %1, off"
               :: "v"(ldsoff), "v"(g) : "memory");
}

__device__ __forceinline__ void wait_async0() {
#if __has_builtin(__builtin_amdgcn_s_wait_asynccnt)
  __builtin_amdgcn_s_wait_asynccnt(0);
#else
  asm volatile("s_wait_asynccnt 0x0" ::: "memory");
#endif
}

// Workgroup barrier ordering ONLY LDS traffic (s_wait_dscnt 0 + split
// barrier). Unlike __syncthreads() it does not drain VMEM load/store
// counters, so global h-stores and seed loads pipeline across LSTM steps.
__device__ __forceinline__ void lds_barrier() {
  asm volatile("" ::: "memory");
#if __has_builtin(__builtin_amdgcn_s_wait_dscnt)
  __builtin_amdgcn_s_wait_dscnt(0);
#else
  asm volatile("s_wait_dscnt 0x0" ::: "memory");
#endif
#if __has_builtin(__builtin_amdgcn_s_barrier_signal) && \
    __has_builtin(__builtin_amdgcn_s_barrier_wait)
  __builtin_amdgcn_s_barrier_signal(-1);
  __builtin_amdgcn_s_barrier_wait(-1);
#else
  asm volatile("s_barrier_signal -1\n\ts_barrier_wait -1" ::: "memory");
#endif
  asm volatile("" ::: "memory");
}

// ---------------------------------------------------------------------------
// Kernel 1: embedding lookup + layer-1 input projections (both directions).
// xg[dir][t][b][g] = emb[x[b, t_in]] . Wih[g,:] + bih[g] + bhh[g]
// Backward direction stores time-reversed projections.
// ---------------------------------------------------------------------------
__global__ __launch_bounds__(256) void k_xg1(
    const int* __restrict__ x, const float* __restrict__ emb,
    const float* __restrict__ wf, const float* __restrict__ bif,
    const float* __restrict__ bhf, const float* __restrict__ wb,
    const float* __restrict__ bib, const float* __restrict__ bhb,
    float* __restrict__ xgF, float* __restrict__ xgB) {
  unsigned idx = blockIdx.x * 256u + threadIdx.x;  // < 2*T*B*G
  int g = idx & (GG - 1);
  unsigned r = idx >> 9;
  int b = r & (BB - 1);
  unsigned r2 = r >> 6;
  int t = r2 & (TT - 1);
  int dir = r2 >> 9;
  int tin = dir ? (TT - 1 - t) : t;
  int tok = x[b * TT + tin];
  const float* W = dir ? wb : wf;
  const float* e = emb + tok * EE;
  float s = dir ? (bib[g] + bhb[g]) : (bif[g] + bhf[g]);
#pragma unroll
  for (int d = 0; d < EE; ++d) s += e[d] * W[g * EE + d];
  float* xg = dir ? xgB : xgF;
  xg[((size_t)t * BB + b) * GG + g] = s;
}

// ---------------------------------------------------------------------------
// Kernel 2: persistent recurrent LSTM. grid.x = 2 (directions), 512 threads
// (16 waves). Whh (f16, B-swizzle) resident in LDS all 512 steps; h state in
// LDS in A-swizzle; c in VGPRs. Wave w: M-tile mt=w&3, hidden slices
// s0=(w>>2), s0+4; per slice it owns gate tiles {s,8+s,16+s,24+s} so the
// cell update is elementwise in registers. kc-outer / gate-inner WMMA chain.
// ---------------------------------------------------------------------------
__global__ __launch_bounds__(512) void k_lstm(
    const float* __restrict__ xgF, const float* __restrict__ xgB,
    const float* __restrict__ whhF, const float* __restrict__ whhB,
    _Float16* __restrict__ outH, float* __restrict__ outF,
    float* __restrict__ hfin) {
  __shared__ v16h whh_sw[32 * 4 * 32];  // [nt][kc][lane] : 128 KB
  __shared__ v16h a_sw[4 * 4 * 32];     // [mt][kc][lane] : 16 KB (h state)

  const int dir = blockIdx.x;
  const float* __restrict__ xg = dir ? xgB : xgF;
  const float* __restrict__ whh = dir ? whhB : whhF;

  const int tid = threadIdx.x;
  const int lane = tid & 31;
  const int w = tid >> 5;
  const int mt = w & 3;
  const int s0 = w >> 2;
  const int hi = lane >> 4;
  const int l16 = lane & 15;

  _Float16* whh_h = (_Float16*)whh_sw;
  for (int e = tid; e < 32 * 4 * 32 * 16; e += 512) {
    int nt = e >> 11, kc = (e >> 9) & 3, ln = (e >> 4) & 31, j = e & 15;
    int g = nt * 16 + (ln & 15);
    int k = kc * 32 + akidx(ln >> 4, j);
    whh_h[e] = (_Float16)whh[g * HH + k];
  }
  _Float16* a_h = (_Float16*)a_sw;
  for (int e = tid; e < 4 * 4 * 32 * 16; e += 512) a_h[e] = (_Float16)0.0f;
  lds_barrier();

  v8f cc[2];
#pragma unroll
  for (int si = 0; si < 2; ++si)
#pragma unroll
    for (int r = 0; r < 8; ++r) cc[si][r] = 0.0f;

#pragma unroll 1
  for (int t = 0; t < TT; ++t) {
    // A fragments (h state): 2x ds_load_b128 each.
    v16h Af[4];
#pragma unroll
    for (int kc = 0; kc < 4; ++kc) Af[kc] = a_sw[(mt * 4 + kc) * 32 + lane];

    // Seed all 8 gate accumulators from xg (issued early, deep in flight;
    // NOT drained at the barrier -- consumed at first WMMA use).
    const float* xgt = xg + (size_t)t * BB * GG;
    v8f g8[2][4];
#pragma unroll
    for (int si = 0; si < 2; ++si)
#pragma unroll
      for (int gi = 0; gi < 4; ++gi) {
        const int nt = gi * 8 + s0 + si * 4;
#pragma unroll
        for (int r = 0; r < 8; ++r)
          g8[si][gi][r] = xgt[(size_t)(mt * 16 + hi * 8 + r) * GG + nt * 16 + l16];
      }

    // Prefetch next step's seed footprint (lane-spread rows, 16 lines per prefetch).
    if (t + 1 < TT) {
      const float* xgn = xg + (size_t)(t + 1) * BB * GG;
#pragma unroll
      for (int si = 0; si < 2; ++si)
#pragma unroll
        for (int gi = 0; gi < 4; ++gi) {
          const int nt = gi * 8 + s0 + si * 4;
          __builtin_prefetch(&xgn[(size_t)(mt * 16 + hi * 8 + (lane & 7)) * GG + nt * 16]);
        }
    }
    lds_barrier();  // everyone captured h before it gets overwritten

    const int tt_out = dir ? (TT - 1 - t) : t;
#pragma unroll
    for (int si = 0; si < 2; ++si) {
      const int s = s0 + si * 4;
#pragma unroll
      for (int kc = 0; kc < 4; ++kc) {
        v16h Bv[4];
#pragma unroll
        for (int gi = 0; gi < 4; ++gi)
          Bv[gi] = whh_sw[((gi * 8 + s) * 4 + kc) * 32 + lane];
#pragma unroll
        for (int gi = 0; gi < 4; ++gi)
          g8[si][gi] = wmma16(Af[kc], Bv[gi], g8[si][gi]);
      }
      // Cell update, elementwise in registers (gate order i,f,g,o).
      v8f hnew;
#pragma unroll
      for (int r = 0; r < 8; ++r) {
        float iv = sigm(g8[si][0][r]);
        float fv = sigm(g8[si][1][r]);
        float gv = fast_tanh(g8[si][2][r]);
        float ov = sigm(g8[si][3][r]);
        float cv = fv * cc[si][r] + iv * gv;
        cc[si][r] = cv;
        hnew[r] = ov * fast_tanh(cv);
      }
      // Write h: global (f16 + f32) and back into the A-swizzled LDS state.
#pragma unroll
      for (int r = 0; r < 8; ++r) {
        int m = hi * 8 + r;
        int brow = mt * 16 + m;
        int hcol = s * 16 + l16;
        _Float16 hv = (_Float16)hnew[r];
        size_t o = ((size_t)brow * TT + tt_out) * DD + dir * HH + hcol;
        outH[o] = hv;
        outF[o] = hnew[r];
        int kc2 = hcol >> 5, Kp = hcol & 31;
        int hi2 = (Kp >> 3) & 1;
        int j2 = (Kp & 7) | ((Kp & 16) >> 1);
        a_h[((mt * 4 + kc2) * 32 + (hi2 * 16 + m)) * 16 + j2] = hv;
      }
    }
    lds_barrier();
  }

  // Final hidden state -> hfin[dir][b][k] (f32) for the attention reshape.
  for (int e = tid; e < 4 * 4 * 32 * 16; e += 512) {
    int mte = e >> 11, kce = (e >> 9) & 3, ln = (e >> 4) & 31, j = e & 15;
    int row = mte * 16 + (ln & 15);
    int k = kce * 32 + akidx(ln >> 4, j);
    hfin[((size_t)dir * BB + row) * HH + k] = (float)a_h[e];
  }
}

// ---------------------------------------------------------------------------
// Kernel 3: prefix-attention scores. S[b][i][t] = (out[b,i].out[b,t])/max(i,1)
// Async global->LDS staging (pair copies), 8 waves x 2 tiles, K = 256.
// ---------------------------------------------------------------------------
__global__ __launch_bounds__(256) void k_scores(const _Float16* __restrict__ outh,
                                                float* __restrict__ S) {
  __shared__ v16h qs[4 * 8 * 32];  // 32 KB A-swizzled
  __shared__ v16h ks[4 * 8 * 32];  // 32 KB B-swizzled
  const int b = blockIdx.x, i0 = blockIdx.y * 64, t0 = blockIdx.z * 64;
  const int tid = threadIdx.x, lane = tid & 31, w = tid >> 5;
  _Float16* qh = (_Float16*)qs;
  _Float16* kh = (_Float16*)ks;
  for (int e2 = tid; e2 < 4 * 8 * 32 * 8; e2 += 256) {  // f16 pairs
    int j = (e2 & 7) * 2;
    int ln = (e2 >> 3) & 31, kc = (e2 >> 8) & 7, mt = e2 >> 11;
    int hie = ln >> 4;
    int row = i0 + mt * 16 + (ln & 15);
    int dA = kc * 32 + akidx(hie, j);
    async_g2l_b32(&outh[((size_t)b * TT + row) * DD + dA], &qh[e2 * 2]);
    int trow = t0 + mt * 16 + (ln & 15);
    int dB = kc * 32 + akidx(hie, j);
    async_g2l_b32(&outh[((size_t)b * TT + trow) * DD + dB], &kh[e2 * 2]);
  }
  wait_async0();
  __syncthreads();
#pragma unroll
  for (int ti = 0; ti < 2; ++ti) {
    int tile = w * 2 + ti, mt = tile & 3, nt = tile >> 2;
    v8f acc = {};
#pragma unroll
    for (int kc = 0; kc < 8; ++kc)
      acc = wmma16(qs[(mt * 8 + kc) * 32 + lane], ks[(nt * 8 + kc) * 32 + lane], acc);
#pragma unroll
    for (int r = 0; r < 8; ++r) {
      int m = (lane >> 4) * 8 + r, n = lane & 15;
      int i = i0 + mt * 16 + m, t = t0 + nt * 16 + n;
      float sc = 1.0f / (float)(i > 1 ? i : 1);
      S[((size_t)b * TT + i) * TT + t] = acc[r] * sc;
    }
  }
}

// ---------------------------------------------------------------------------
// Kernel 4: softmax over the BATCH axis (64 values) per (i,t), strict lower-
// triangular mask. lane <-> t so every b-iteration is a coalesced line.
// ---------------------------------------------------------------------------
__global__ __launch_bounds__(256) void k_softmax_b(const float* __restrict__ S,
                                                   _Float16* __restrict__ Wh) {
  int idx = blockIdx.x * 256 + threadIdx.x;  // T*T threads
  int i = idx >> 9, t = idx & (TT - 1);
  size_t base = (size_t)i * TT + t;
  const size_t bs = (size_t)TT * TT;
  float m = -1e30f;
  for (int b = 0; b < BB; ++b) m = fmaxf(m, S[base + b * bs]);
  float ssum = 0.0f;
  for (int b = 0; b < BB; ++b) ssum += __expf(S[base + b * bs] - m);
  float inv = 1.0f / ssum;
  bool keep = (t < i);
  for (int b = 0; b < BB; ++b) {
    float e = __expf(S[base + b * bs] - m) * inv;
    Wh[base + b * bs] = keep ? (_Float16)e : (_Float16)0.0f;
  }
}

// ---------------------------------------------------------------------------
// Kernel 5: att[b][i][d] = sum_t W[b][i][t] * out[b][t][d].  K=512 staged in
// 4 chunks of 128 through LDS with async copies.
// ---------------------------------------------------------------------------
__global__ __launch_bounds__(256) void k_apply(const _Float16* __restrict__ Wh,
                                               const _Float16* __restrict__ outh,
                                               _Float16* __restrict__ atth) {
  __shared__ v16h As[4 * 4 * 32];  // 16 KB
  __shared__ v16h Bs[4 * 4 * 32];  // 16 KB
  const int b = blockIdx.x, i0 = blockIdx.y * 64, d0 = blockIdx.z * 64;
  const int tid = threadIdx.x, lane = tid & 31, w = tid >> 5;
  _Float16* ah = (_Float16*)As;
  _Float16* bh = (_Float16*)Bs;
  v8f acc[2];
#pragma unroll
  for (int ti = 0; ti < 2; ++ti)
#pragma unroll
    for (int r = 0; r < 8; ++r) acc[ti][r] = 0.0f;

  for (int kst = 0; kst < 4; ++kst) {
    if (kst) __syncthreads();
    for (int e2 = tid; e2 < 4 * 4 * 32 * 8; e2 += 256) {  // f16 pairs
      int j = (e2 & 7) * 2;
      int ln = (e2 >> 3) & 31, kc = (e2 >> 8) & 3, mt = e2 >> 10;
      int hie = ln >> 4;
      int i = i0 + mt * 16 + (ln & 15);
      int tcol = kst * 128 + kc * 32 + akidx(hie, j);
      async_g2l_b32(&Wh[((size_t)b * TT + i) * TT + tcol], &ah[e2 * 2]);
      int d = d0 + mt * 16 + (ln & 15);
      int tB = kst * 128 + kc * 32 + akidx(hie, j);
      bh[e2 * 2] = outh[((size_t)b * TT + tB) * DD + d];
      bh[e2 * 2 + 1] = outh[((size_t)b * TT + tB + 1) * DD + d];
    }
    wait_async0();
    __syncthreads();
#pragma unroll
    for (int ti = 0; ti < 2; ++ti) {
      int tile = w * 2 + ti, mt = tile & 3, nt = tile >> 2;
#pragma unroll
      for (int kc = 0; kc < 4; ++kc)
        acc[ti] = wmma16(As[(mt * 4 + kc) * 32 + lane],
                         Bs[(nt * 4 + kc) * 32 + lane], acc[ti]);
    }
  }
#pragma unroll
  for (int ti = 0; ti < 2; ++ti) {
    int tile = w * 2 + ti, mt = tile & 3, nt = tile >> 2;
#pragma unroll
    for (int r = 0; r < 8; ++r) {
      int i = i0 + mt * 16 + (lane >> 4) * 8 + r;
      int d = d0 + nt * 16 + (lane & 15);
      atth[((size_t)b * TT + i) * DD + d] = (_Float16)acc[ti][r];
    }
  }
}

// ---------------------------------------------------------------------------
// Kernel 6: layer-2 input projection, xg2[t][b][g] = att[b][tin] . Wih2[g,:]
// + biases. A-side async staging; B-side converts f32 weights to f16.
// ---------------------------------------------------------------------------
__global__ __launch_bounds__(256) void k_xg2(const _Float16* __restrict__ atth,
                                             const float* __restrict__ wih,
                                             const float* __restrict__ bi,
                                             const float* __restrict__ bh,
                                             float* __restrict__ xg2, int dir) {
  __shared__ v16h As[4 * 8 * 32];  // 32 KB att rows (b as M)
  __shared__ v16h Bs[4 * 8 * 32];  // 32 KB Wih^T
  const int tq = blockIdx.x, g0 = blockIdx.y * 64;
  const int tin = dir ? (TT - 1 - tq) : tq;
  const int tid = threadIdx.x, lane = tid & 31, w = tid >> 5;
  _Float16* ah = (_Float16*)As;
  _Float16* bhh_ = (_Float16*)Bs;
  for (int e2 = tid; e2 < 4 * 8 * 32 * 8; e2 += 256) {  // A: f16 pairs, async
    int j = (e2 & 7) * 2;
    int ln = (e2 >> 3) & 31, kc = (e2 >> 8) & 7, mt = e2 >> 11;
    int hie = ln >> 4;
    int brow = mt * 16 + (ln & 15);
    int dA = kc * 32 + akidx(hie, j);
    async_g2l_b32(&atth[((size_t)brow * TT + tin) * DD + dA], &ah[e2 * 2]);
  }
  for (int e = tid; e < 4 * 8 * 32 * 16; e += 256) {  // B: f32 -> f16 convert
    int mt = e >> 12, kc = (e >> 9) & 7, ln = (e >> 4) & 31, j = e & 15;
    int hie = ln >> 4;
    int g = g0 + mt * 16 + (ln & 15);
    int dB = kc * 32 + akidx(hie, j);
    bhh_[e] = (_Float16)wih[(size_t)g * DD + dB];
  }
  wait_async0();
  __syncthreads();
#pragma unroll
  for (int ti = 0; ti < 2; ++ti) {
    int tile = w * 2 + ti, mt = tile & 3, nt = tile >> 2;
    v8f acc = {};
#pragma unroll
    for (int kc = 0; kc < 8; ++kc)
      acc = wmma16(As[(mt * 8 + kc) * 32 + lane], Bs[(nt * 8 + kc) * 32 + lane], acc);
#pragma unroll
    for (int r = 0; r < 8; ++r) {
      int bq = mt * 16 + (lane >> 4) * 8 + r;
      int g = g0 + nt * 16 + (lane & 15);
      xg2[((size_t)tq * BB + bq) * GG + g] = acc[r] + bi[g] + bh[g];
    }
  }
}

// ---------------------------------------------------------------------------
// Full-attention helpers (tiny). hidden = raw reshape of h_n[2,B,H] -> [B,2H].
// ---------------------------------------------------------------------------
__global__ __launch_bounds__(256) void k_attn_score(const float* __restrict__ outf,
                                                    const float* __restrict__ hfl,
                                                    float* __restrict__ sc) {
  int idx = blockIdx.x * 256 + threadIdx.x;  // B*T
  int b = idx >> 9, t = idx & (TT - 1);
  float s = 0.0f;
  const float* o = outf + ((size_t)b * TT + t) * DD;
  const float* h = hfl + (size_t)b * DD;  // torch-faithful flat reshape
#pragma unroll 8
  for (int d = 0; d < DD; ++d) s += o[d] * h[d];
  sc[b * TT + t] = s * (1.0f / (float)TT);
}

__global__ __launch_bounds__(256) void k_attn_soft(float* __restrict__ sc) {
  int t = blockIdx.x * 256 + threadIdx.x;
  if (t >= TT) return;
  float m = -1e30f;
  for (int b = 0; b < BB; ++b) m = fmaxf(m, sc[b * TT + t]);
  float s = 0.0f;
  for (int b = 0; b < BB; ++b) s += __expf(sc[b * TT + t] - m);
  float inv = 1.0f / s;
  for (int b = 0; b < BB; ++b) sc[b * TT + t] = __expf(sc[b * TT + t] - m) * inv;
}

__global__ __launch_bounds__(256) void k_attn_apply_row(
    const float* __restrict__ outf, const float* __restrict__ sc,
    _Float16* __restrict__ atth) {
  int idx = blockIdx.x * 256 + threadIdx.x;  // B*D
  int b = idx >> 8, d = idx & (DD - 1);
  float s = 0.0f;
  for (int t = 0; t < TT; ++t) s += sc[b * TT + t] * outf[((size_t)b * TT + t) * DD + d];
  atth[((size_t)b * TT + (TT - 1)) * DD + d] = (_Float16)s;
}

__global__ __launch_bounds__(256) void k_attn_apply_a2(
    const float* __restrict__ outf, const float* __restrict__ sc,
    float* __restrict__ a2) {
  int idx = blockIdx.x * 256 + threadIdx.x;
  int b = idx >> 8, d = idx & (DD - 1);
  float s = 0.0f;
  for (int t = 0; t < TT; ++t) s += sc[b * TT + t] * outf[((size_t)b * TT + t) * DD + d];
  a2[(size_t)b * DD + d] = s;
}

__global__ __launch_bounds__(64) void k_final(const float* __restrict__ a2,
                                              const float* __restrict__ lw,
                                              const float* __restrict__ lb,
                                              float* __restrict__ out) {
  int b = threadIdx.x;
  float s = lb[0];
#pragma unroll 8
  for (int d = 0; d < DD; ++d) s += a2[(size_t)b * DD + d] * lw[d];
  out[b] = sigm(s);
}

// ---------------------------------------------------------------------------
extern "C" void kernel_launch(void* const* d_in, const int* in_sizes, int n_in,
                              void* d_out, int out_size, void* d_ws, size_t ws_size,
                              hipStream_t stream) {
  (void)in_sizes; (void)n_in; (void)out_size; (void)ws_size;
  const int*   x    = (const int*)d_in[0];
  const float* emb  = (const float*)d_in[1];
  const float* w1f  = (const float*)d_in[2];
  const float* u1f  = (const float*)d_in[3];
  const float* bi1f = (const float*)d_in[4];
  const float* bh1f = (const float*)d_in[5];
  const float* w1b  = (const float*)d_in[6];
  const float* u1b  = (const float*)d_in[7];
  const float* bi1b = (const float*)d_in[8];
  const float* bh1b = (const float*)d_in[9];
  const float* w2f  = (const float*)d_in[10];
  const float* u2f  = (const float*)d_in[11];
  const float* bi2f = (const float*)d_in[12];
  const float* bh2f = (const float*)d_in[13];
  const float* w2b  = (const float*)d_in[14];
  const float* u2b  = (const float*)d_in[15];
  const float* bi2b = (const float*)d_in[16];
  const float* bh2b = (const float*)d_in[17];
  const float* linW = (const float*)d_in[18];
  const float* linb = (const float*)d_in[19];

  char* ws = (char*)d_ws;
  constexpr size_t SZ_XG  = (size_t)TT * BB * GG * 4;      // 64 MB
  constexpr size_t SZ_OH  = (size_t)BB * TT * DD * 2;      // 16 MB
  constexpr size_t SZ_OF  = (size_t)BB * TT * DD * 4;      // 32 MB
  constexpr size_t SZ_S   = (size_t)BB * TT * TT * 4;      // 64 MB
  constexpr size_t SZ_WH  = (size_t)BB * TT * TT * 2;      // 32 MB
  constexpr size_t OFF_XGF = 0;
  constexpr size_t OFF_XGB = OFF_XGF + SZ_XG;
  constexpr size_t OFF_OH  = OFF_XGB + SZ_XG;              // out1 then out2
  constexpr size_t OFF_OF  = OFF_OH + SZ_OH;
  constexpr size_t OFF_S   = OFF_OF + SZ_OF;               // S, then att (alias)
  constexpr size_t OFF_WH  = OFF_S + SZ_S;
  constexpr size_t OFF_H1  = OFF_WH + SZ_WH;
  constexpr size_t OFF_H2  = OFF_H1 + 2 * BB * HH * 4;
  constexpr size_t OFF_SC  = OFF_H2 + 2 * BB * HH * 4;
  constexpr size_t OFF_A2  = OFF_SC + (size_t)BB * TT * 4;

  float*     xgF  = (float*)(ws + OFF_XGF);
  float*     xgB  = (float*)(ws + OFF_XGB);
  _Float16*  outh = (_Float16*)(ws + OFF_OH);
  float*     outf = (float*)(ws + OFF_OF);
  float*     S    = (float*)(ws + OFF_S);
  _Float16*  atth = (_Float16*)(ws + OFF_S);   // aliases S (S dead by then)
  _Float16*  Wh   = (_Float16*)(ws + OFF_WH);
  float*     h1   = (float*)(ws + OFF_H1);
  float*     h2   = (float*)(ws + OFF_H2);
  float*     scb  = (float*)(ws + OFF_SC);
  float*     a2   = (float*)(ws + OFF_A2);

  // 1) embedding + layer-1 input projections (both directions)
  k_xg1<<<(2u * TT * BB * GG) / 256, 256, 0, stream>>>(
      x, emb, w1f, bi1f, bh1f, w1b, bi1b, bh1b, xgF, xgB);
  // 2) layer-1 BiLSTM (persistent, 1 workgroup / direction)
  k_lstm<<<2, 512, 0, stream>>>(xgF, xgB, u1f, u1b, outh, outf, h1);
  // 3) prefix attention: scores -> batch softmax+mask -> apply
  k_scores<<<dim3(BB, TT / 64, TT / 64), 256, 0, stream>>>(outh, S);
  k_softmax_b<<<(TT * TT) / 256, 256, 0, stream>>>(S, Wh);
  k_apply<<<dim3(BB, TT / 64, DD / 64), 256, 0, stream>>>(Wh, outh, atth);
  // 4) full attention on out1 overwrites att[:, T-1]
  k_attn_score<<<(BB * TT) / 256, 256, 0, stream>>>(outf, h1, scb);
  k_attn_soft<<<2, 256, 0, stream>>>(scb);
  k_attn_apply_row<<<(BB * DD) / 256, 256, 0, stream>>>(outf, scb, atth);
  // 5) layer-2 input projections (WMMA GEMM, reuses xg region)
  k_xg2<<<dim3(TT, GG / 64), 256, 0, stream>>>(atth, w2f, bi2f, bh2f, xgF, 0);
  k_xg2<<<dim3(TT, GG / 64), 256, 0, stream>>>(atth, w2b, bi2b, bh2b, xgB, 1);
  // 6) layer-2 BiLSTM (reuses out buffers)
  k_lstm<<<2, 512, 0, stream>>>(xgF, xgB, u2f, u2b, outh, outf, h2);
  // 7) full attention on out2 + final linear + sigmoid
  k_attn_score<<<(BB * TT) / 256, 256, 0, stream>>>(outf, h2, scb);
  k_attn_soft<<<2, 256, 0, stream>>>(scb);
  k_attn_apply_a2<<<(BB * DD) / 256, 256, 0, stream>>>(outf, scb, a2);
  k_final<<<1, 64, 0, stream>>>(a2, linW, linb, (float*)d_out);
}
